// FactorGraphNeuralNetwork_58222576664744
// MI455X (gfx1250) — matrix operand order repaired and checked by reading, not compiled, not run on hardware
//
#include <hip/hip_runtime.h>
#include <hip/hip_bf16.h>
#include <cstdint>

typedef __attribute__((ext_vector_type(16))) _Float16 v16h;
typedef __attribute__((ext_vector_type(8)))  _Float16 v8h;
typedef __attribute__((ext_vector_type(8)))  float    v8f;
typedef __attribute__((ext_vector_type(4)))  unsigned int u32x4;
typedef __attribute__((ext_vector_type(8)))  int      i32x8;
typedef __attribute__((ext_vector_type(4)))  int      i32x4;

#define HID   64
#define NB    32
#define NM    256
#define NK    128
#define NTOK  (NB*NM*NK)          /* 1048576 tokens */
#define EPSLN 1e-5f

/* ---- workspace layout (units: halves) ---- */
#define O_ENCW2  0                 /* 64x64   packed -> 4096   */
#define O_LW1    4096              /* 4x 64x64        -> 16384 */
#define O_LW2    20480             /* 4x 64x64        -> 16384 */
#define O_GWIH   36864             /* 4x 192x64       -> 49152 */
#define O_GWHH   86016             /* 4x 192x64       -> 49152 */
#define O_HEADW1 135168            /* 64x128          -> 8192  */
#define O_HID16  143360            /* new_hidden f16  -> 2048  */
#define O_X      145408            /* x tensor f16: NTOK*HID   */
#define H_TOTAL  (O_X + (size_t)NTOK*HID)
/* f32 scratch region (units: floats), placed after the halves region */
#define F_PART 0                   /* NB*64*HID partial maxes */
#define F_GMAX (NB*64*HID)
#define F_NEWH (F_GMAX + NB*HID)

/* ============================ device helpers ============================ */

__device__ inline float sigm(float x)   { return 1.0f / (1.0f + __expf(-x)); }
__device__ inline float tanh_f(float x) { float e = __expf(-2.0f*x); return (1.0f - e)/(1.0f + e); }

__device__ inline unsigned lds_off(const void* p) {   /* low 32 bits of flat LDS addr = LDS byte offset */
  return (unsigned)(uintptr_t)p;
}

/* ---- Tensor Data Mover: build D# (groups 0/1 per CDNA5 ISA) and issue ----
   2D tile: tile_d0 elements per row, tile_d1 rows (0 => 1D), global row stride stride0 (elements),
   data_size = 2 bytes (f16). LDS side compacted unless padding enabled (raw field codes). */
__device__ inline void tdm_issue(unsigned ldsAddr, unsigned long ga,
                                 unsigned tile_d0, unsigned tile_d1,
                                 unsigned long stride0,
                                 unsigned pad_interval_code, unsigned pad_amount_code,
                                 bool pad_en, bool is_store) {
  u32x4 g0;
  g0[0] = 1u;                                            /* count=1, user mode */
  g0[1] = ldsAddr;                                       /* lds_addr (bytes) */
  g0[2] = (unsigned)(ga & 0xffffffffu);                  /* global_addr[31:0] */
  g0[3] = (unsigned)((ga >> 32) & 0x01ffffffu) | (2u << 30);  /* addr[56:32] | type=2 */
  const unsigned td0 = 0x40000000u, td1 = 0x40000000u;   /* huge tensor dims: no OOB clipping */
  i32x8 g1;
  g1[0] = (int)((1u << 16) |                             /* data_size=1 -> 2 bytes */
                (pad_en ? (1u << 20) : 0u) |
                (pad_interval_code << 22) | (pad_amount_code << 25));
  g1[1] = (int)((td0 & 0xffffu) << 16);                  /* ab_addr=0 | tensor_dim0[15:0] */
  g1[2] = (int)((td0 >> 16) | ((td1 & 0xffffu) << 16));  /* td0[31:16] | td1[15:0] */
  g1[3] = (int)((td1 >> 16) | (tile_d0 << 16));          /* td1[31:16] | tile_dim0 */
  g1[4] = (int)(tile_d1 & 0xffffu);                      /* tile_dim1 | tile_dim2=0 */
  g1[5] = (int)(unsigned)(stride0 & 0xffffffffu);        /* tensor_dim0_stride[31:0] */
  g1[6] = (int)(unsigned)((stride0 >> 32) & 0xffffu);    /* stride0[47:32] | stride1[15:0]=0 */
  g1[7] = 0;
  i32x4 z4; z4[0] = 0; z4[1] = 0; z4[2] = 0; z4[3] = 0;  /* groups 2/3 unused (<=2D) */
  i32x8 z8;
#pragma unroll
  for (int i = 0; i < 8; ++i) z8[i] = 0;
  if (is_store) __builtin_amdgcn_tensor_store_from_lds(g0, g1, z4, z4, z8, 0);
  else          __builtin_amdgcn_tensor_load_to_lds  (g0, g1, z4, z4, z8, 0);
}

/* broadcast bias into C layout: element (m,n) lives in lane n+16*(m>=8), vgpr m%8 */
__device__ inline v8f acc_bias(const float* __restrict__ b, int n0) {
  float bv = b[n0 + (threadIdx.x & 15)];
  v8f acc;
#pragma unroll
  for (int i = 0; i < 8; ++i) acc[i] = bv;
  return acc;
}

/* 16(tokens) x 16(outs) tile, K accumulated in steps of 32 via v_wmma_f32_16x16x32_f16. */
__device__ inline v8f gemm16(const _Float16* As, int TW, int t0,
                             const _Float16* __restrict__ Bp, int nt, int KT,
                             v8f acc) {
  const int lane = threadIdx.x & 31;
  const int row  = lane & 15;
  const int kh   = lane >> 4;
#pragma unroll
  for (int kt = 0; kt < KT; ++kt) {
    const v8h* pa = (const v8h*)(As + (t0 + row)*TW + kt*32 + kh*8);
    v8h lo = pa[0];
    v8h hi = pa[2];                       /* +16 halves */
    v16h a = __builtin_shufflevector(lo, hi, 0,1,2,3,4,5,6,7,8,9,10,11,12,13,14,15);
    v16h b = *(const v16h*)(Bp + ((size_t)((nt*KT + kt)*32 + lane) << 4));
    acc = __builtin_amdgcn_wmma_f32_16x16x32_f16(false, a, false, b,
                                                 (short)0, acc, false, false);
  }
  return acc;
}

__device__ inline void store_tile(float* S, int TW, int t0, int n0, v8f acc) {
  const int lane = threadIdx.x & 31;
  const int n  = n0 + (lane & 15);
  const int tb = t0 + ((lane >> 4) << 3);
#pragma unroll
  for (int v = 0; v < 8; ++v) S[(tb + v)*TW + n] = acc[v];
}

/* ============================ weight packing ============================ */
__global__ void k_prep(const float* __restrict__ enc_w2,
                       const float* __restrict__ l_w1,  const float* __restrict__ l_w2,
                       const float* __restrict__ l_gwih,const float* __restrict__ l_gwhh,
                       const float* __restrict__ head_w1,
                       _Float16* __restrict__ wsh) {
  int idx = blockIdx.x * blockDim.x + threadIdx.x;
  const float* W; _Float16* dst; int Kin, local;
  if (idx < 4096)        { W = enc_w2; dst = wsh + O_ENCW2; Kin = 64;  local = idx; }
  else if (idx < 20480)  { int r = idx - 4096;  int t = r >> 12;
                           W = l_w1 + t*4096;  dst = wsh + O_LW1  + t*4096;  Kin = 64;  local = r & 4095; }
  else if (idx < 36864)  { int r = idx - 20480; int t = r >> 12;
                           W = l_w2 + t*4096;  dst = wsh + O_LW2  + t*4096;  Kin = 64;  local = r & 4095; }
  else if (idx < 86016)  { int r = idx - 36864; int t = r / 12288;
                           W = l_gwih + t*12288; dst = wsh + O_GWIH + t*12288; Kin = 64; local = r - t*12288; }
  else if (idx < 135168) { int r = idx - 86016; int t = r / 12288;
                           W = l_gwhh + t*12288; dst = wsh + O_GWHH + t*12288; Kin = 64; local = r - t*12288; }
  else if (idx < 143360) { W = head_w1; dst = wsh + O_HEADW1; Kin = 128; local = idx - 135168; }
  else return;
  int e = local & 15, L = (local >> 4) & 31, rest = local >> 9;
  int KT = Kin >> 5;
  int kt = rest % KT, nt = rest / KT;
  int n = nt*16 + (L & 15);
  int k = kt*32 + ((e >> 3) << 4) + ((L >> 4) << 3) + (e & 7);
  dst[local] = (_Float16)W[n*Kin + k];
}

/* ============================ encoder ============================ */
__global__ void __launch_bounds__(256) k_encoder(
    const float* __restrict__ inp, const float* __restrict__ w1,
    const float* __restrict__ b1,  const float* __restrict__ g,
    const float* __restrict__ beta,const _Float16* __restrict__ w2p,
    const float* __restrict__ b2,  _Float16* __restrict__ xg) {
  __shared__ float s_in[128][4];
  __shared__ float s_h [128][HID];
  __shared__ __align__(16) _Float16 s_a[128][HID];
  const int tid = threadIdx.x;
  const long t0 = (long)blockIdx.x * 128;

  for (int i = tid; i < 128*4; i += 256) s_in[i >> 2][i & 3] = inp[t0*4 + i];
  __syncthreads();
  for (int o = tid; o < 128*HID; o += 256) {   /* Linear 4->64 (tiny, VALU) */
    int t = o >> 6, j = o & 63;
    float acc = b1[j];
#pragma unroll
    for (int c = 0; c < 4; ++c) acc += s_in[t][c] * w1[j*4 + c];
    s_h[t][j] = acc;
  }
  __syncthreads();
  if (tid < 128) {                              /* LayerNorm + ReLU -> f16 */
    float mu = 0.f, var = 0.f;
    for (int j = 0; j < HID; ++j) mu += s_h[tid][j];
    mu *= (1.0f/HID);
    for (int j = 0; j < HID; ++j) { float d = s_h[tid][j] - mu; var += d*d; }
    float rs = __frsqrt_rn(var*(1.0f/HID) + EPSLN);
    for (int j = 0; j < HID; ++j) {
      float v = (s_h[tid][j] - mu)*rs*g[j] + beta[j];
      s_a[tid][j] = (_Float16)fmaxf(v, 0.0f);
    }
  }
  __syncthreads();
  const int wave = tid >> 5, lane = tid & 31;
  _Float16* s_o = (_Float16*)&s_h[0][0];        /* reuse f32 scratch as f16 out tile */
  const int tt = wave * 16;                     /* Linear 64->64 via WMMA */
#pragma unroll
  for (int nt = 0; nt < 4; ++nt) {
    v8f acc = acc_bias(b2, nt*16);
    acc = gemm16(&s_a[0][0], HID, tt, w2p, nt, 2, acc);
    int n  = nt*16 + (lane & 15);
    int tb = tt + ((lane >> 4) << 3);
#pragma unroll
    for (int v = 0; v < 8; ++v)
      s_o[(tb + v)*HID + n] = (_Float16)acc[v];
  }
  __syncthreads();
  if (tid < 32) {                               /* bulk DMA: LDS -> global (contiguous 8192 halves) */
    tdm_issue(lds_off(s_o), (unsigned long)(xg + t0*HID),
              128*HID, 0, 128*HID, 0, 0, false, true);
    __builtin_amdgcn_s_wait_tensorcnt(0);
  }
}

/* ============================ fused layer pass ============================ */
template<int T, int TSTRIDE>
__global__ void __launch_bounds__(256) k_pass(
    _Float16* __restrict__ xg,
    const _Float16* __restrict__ w1p, const _Float16* __restrict__ w2p,
    const _Float16* __restrict__ wihp,const _Float16* __restrict__ whhp,
    const float* __restrict__ b1,  const float* __restrict__ g,
    const float* __restrict__ beta,const float* __restrict__ b2,
    const float* __restrict__ bih, const float* __restrict__ bhh) {
  __shared__ __align__(16) _Float16 s_x[T][HID];   /* h (original x), f16 */
  __shared__ __align__(16) _Float16 s_a[T][HID];   /* activation staging, f16 */
  __shared__ __align__(16) float s_f[T][HID];      /* f32 GEMM scratch / f16 out tile */
  __shared__ float s_max[HID];
  __shared__ float s_pmax[4][HID];

  const int tid = threadIdx.x;
  const int bid = blockIdx.x;
  long base;
  if (TSTRIDE == 1) base = (long)bid * T;
  else              base = (long)(bid / NK) * (NM*NK) + (bid % NK);

  __builtin_prefetch(w1p,  0, 0);
  __builtin_prefetch(w2p,  0, 0);
  __builtin_prefetch(wihp, 0, 0);
  __builtin_prefetch(whhp, 0, 0);

  /* Tensor Data Mover: DMA the block's x tile into LDS.
     row pass: 1D contiguous T*HID halves; col pass: 2D 64 x T rows, stride NK*HID elems. */
  if (tid < 32) {
    if (TSTRIDE == 1)
      tdm_issue(lds_off(&s_x[0][0]), (unsigned long)(xg + base*HID),
                (unsigned)(T*HID), 0u, (unsigned long)(T*HID), 0, 0, false, false);
    else
      tdm_issue(lds_off(&s_x[0][0]), (unsigned long)(xg + base*HID),
                (unsigned)HID, (unsigned)T, (unsigned long)TSTRIDE*HID, 0, 0, false, false);
    __builtin_amdgcn_s_wait_tensorcnt(0);
  }
  __syncthreads();

  { /* per-feature max over the T tokens */
    int j = tid & 63, p = tid >> 6;
    float m = -3.0e38f;
    for (int t = p; t < T; t += 4) m = fmaxf(m, (float)s_x[t][j]);
    s_pmax[p][j] = m;
  }
  __syncthreads();
  if (tid < HID)
    s_max[tid] = fmaxf(fmaxf(s_pmax[0][tid], s_pmax[1][tid]),
                       fmaxf(s_pmax[2][tid], s_pmax[3][tid]));
  __syncthreads();
  for (int i = tid; i < T*HID; i += 256) {
    int t = i >> 6, j = i & 63;
    s_a[t][j] = (_Float16)((float)s_x[t][j] - s_max[j]);
  }
  __syncthreads();

  const int wave = tid >> 5, lane = tid & 31;

  /* GEMM1: s_f = (x - max) @ w1^T + b1 */
  for (int tile = wave; tile < T/16; tile += 8) {
    int t0 = tile*16;
#pragma unroll
    for (int nt = 0; nt < 4; ++nt) {
      v8f acc = acc_bias(b1, nt*16);
      acc = gemm16(&s_a[0][0], HID, t0, w1p, nt, 2, acc);
      store_tile(&s_f[0][0], HID, t0, nt*16, acc);
    }
  }
  __syncthreads();
  /* LayerNorm + ReLU -> s_a (f16) */
  for (int t = tid; t < T; t += 256) {
    float mu = 0.f, var = 0.f;
    for (int j = 0; j < HID; ++j) mu += s_f[t][j];
    mu *= (1.0f/HID);
    for (int j = 0; j < HID; ++j) { float d = s_f[t][j] - mu; var += d*d; }
    float rs = __frsqrt_rn(var*(1.0f/HID) + EPSLN);
    for (int j = 0; j < HID; ++j) {
      float v = (s_f[t][j] - mu)*rs*g[j] + beta[j];
      s_a[t][j] = (_Float16)fmaxf(v, 0.0f);
    }
  }
  __syncthreads();
  /* GEMM2: s_f = relu(ln) @ w2^T + b2  (= upd) */
  for (int tile = wave; tile < T/16; tile += 8) {
    int t0 = tile*16;
#pragma unroll
    for (int nt = 0; nt < 4; ++nt) {
      v8f acc = acc_bias(b2, nt*16);
      acc = gemm16(&s_a[0][0], HID, t0, w2p, nt, 2, acc);
      store_tile(&s_f[0][0], HID, t0, nt*16, acc);
    }
  }
  __syncthreads();
  for (int i = tid; i < T*HID; i += 256) {   /* upd -> f16 */
    int t = i >> 6, j = i & 63;
    s_a[t][j] = (_Float16)s_f[t][j];
  }
  __syncthreads();

  /* fused GRUCell: 6 gate GEMMs (K=64) + epilogue -> f16 out tile in LDS (s_f reused) */
  _Float16* s_o = (_Float16*)&s_f[0][0];
  for (int tile = wave; tile < T/16; tile += 8) {
    int t0 = tile*16;
#pragma unroll
    for (int nt = 0; nt < 4; ++nt) {
      const int j0 = nt*16;
      v8f air = acc_bias(bih, j0);
      v8f aiz = acc_bias(bih, j0 + 64);
      v8f ain = acc_bias(bih, j0 + 128);
      v8f ahr = acc_bias(bhh, j0);
      v8f ahz = acc_bias(bhh, j0 + 64);
      v8f ahn = acc_bias(bhh, j0 + 128);
      air = gemm16(&s_a[0][0], HID, t0, wihp, nt,     2, air);
      aiz = gemm16(&s_a[0][0], HID, t0, wihp, nt + 4, 2, aiz);
      ain = gemm16(&s_a[0][0], HID, t0, wihp, nt + 8, 2, ain);
      ahr = gemm16(&s_x[0][0], HID, t0, whhp, nt,     2, ahr);
      ahz = gemm16(&s_x[0][0], HID, t0, whhp, nt + 4, 2, ahz);
      ahn = gemm16(&s_x[0][0], HID, t0, whhp, nt + 8, 2, ahn);
      const int n  = j0 + (lane & 15);
      const int tb = t0 + ((lane >> 4) << 3);
#pragma unroll
      for (int v = 0; v < 8; ++v) {
        float r  = sigm(air[v] + ahr[v]);
        float z  = sigm(aiz[v] + ahz[v]);
        float nn = tanh_f(ain[v] + r*ahn[v]);
        float xo = (float)s_x[tb + v][n];
        float nx = (1.0f - z)*nn + z*xo;
        s_o[(tb + v)*HID + n] = (_Float16)nx;
      }
    }
  }
  __syncthreads();
  if (tid < 32) {   /* bulk DMA: new x tile LDS -> global */
    if (TSTRIDE == 1)
      tdm_issue(lds_off(s_o), (unsigned long)(xg + base*HID),
                (unsigned)(T*HID), 0u, (unsigned long)(T*HID), 0, 0, false, true);
    else
      tdm_issue(lds_off(s_o), (unsigned long)(xg + base*HID),
                (unsigned)HID, (unsigned)T, (unsigned long)TSTRIDE*HID, 0, 0, false, true);
    __builtin_amdgcn_s_wait_tensorcnt(0);
  }
}

/* ============================ global max (2-stage) ============================ */
__global__ void __launch_bounds__(256) k_gmax1(const _Float16* __restrict__ xg,
                                               float* __restrict__ part) {
  __shared__ float s_p[4][HID];
  const int bid = blockIdx.x;                 /* NB*64 blocks */
  const int b = bid >> 6, blk = bid & 63;
  const long tok0 = (long)b*(NM*NK) + (long)blk*512;
  const int j = threadIdx.x & 63, p = threadIdx.x >> 6;
  float m = -3.0e38f;
  for (int i = p; i < 512; i += 4) m = fmaxf(m, (float)xg[(tok0 + i)*HID + j]);
  s_p[p][j] = m;
  __syncthreads();
  if (threadIdx.x < HID)
    part[bid*HID + threadIdx.x] =
        fmaxf(fmaxf(s_p[0][threadIdx.x], s_p[1][threadIdx.x]),
              fmaxf(s_p[2][threadIdx.x], s_p[3][threadIdx.x]));
}

__global__ void k_gmax2(const float* __restrict__ part, float* __restrict__ gmax) {
  const int b = blockIdx.x, j = threadIdx.x;  /* 64 threads */
  float m = -3.0e38f;
  for (int blk = 0; blk < 64; ++blk) m = fmaxf(m, part[(b*64 + blk)*HID + j]);
  gmax[b*HID + j] = m;
}

/* ============================ top-level GRU (tiny) ============================ */
__global__ void k_topgru(const float* __restrict__ gmax, const float* __restrict__ h0,
                         const float* __restrict__ wih,  const float* __restrict__ whh,
                         const float* __restrict__ bih,  const float* __restrict__ bhh,
                         float* __restrict__ newh_f, _Float16* __restrict__ newh_h,
                         float* __restrict__ out_tail) {
  int idx = blockIdx.x*blockDim.x + threadIdx.x;
  if (idx >= NB*HID) return;
  const int b = idx >> 6, j = idx & 63;
  const float* xf = gmax + b*HID;
  const float* hf = h0   + b*HID;
  float ir = bih[j], iz = bih[j+64], in = bih[j+128];
  float hr = bhh[j], hz = bhh[j+64], hn = bhh[j+128];
  for (int k = 0; k < HID; ++k) {
    float xv = xf[k], hv = hf[k];
    ir += xv*wih[j*HID + k];       iz += xv*wih[(j+64)*HID + k];  in += xv*wih[(j+128)*HID + k];
    hr += hv*whh[j*HID + k];       hz += hv*whh[(j+64)*HID + k];  hn += hv*whh[(j+128)*HID + k];
  }
  float r  = sigm(ir + hr), z = sigm(iz + hz);
  float nn = tanh_f(in + r*hn);
  float nh = (1.0f - z)*nn + z*hf[j];
  newh_f[idx]  = nh;
  newh_h[idx]  = (_Float16)nh;
  out_tail[idx] = nh;
}

/* ============================ head ============================ */
__global__ void __launch_bounds__(256) k_head(
    const _Float16* __restrict__ xg, const _Float16* __restrict__ hid16,
    const _Float16* __restrict__ w1p, const float* __restrict__ b1,
    const float* __restrict__ w2, const float* __restrict__ b2,
    float* __restrict__ out) {
  __shared__ __align__(16) _Float16 s_c[128][2*HID];  /* [x ‖ h] */
  __shared__ float s_h[128][HID];
  const int tid = threadIdx.x;
  const long t0 = (long)blockIdx.x * 128;
  const int b = (int)(t0 >> 15);                      /* 32768 tokens / batch */

  /* TDM load of the x half of s_c: 1D 8192 halves, LDS-padded 32 DWORDs every 32 DWORDs
     so each 64-half payload lands at a 128-half LDS row stride. */
  if (tid < 32) {
    tdm_issue(lds_off(&s_c[0][0]), (unsigned long)(xg + t0*HID),
              128*HID, 0, 128*HID, /*pad_interval 32DW*/4, /*pad 32DW*/31, true, false);
    __builtin_amdgcn_s_wait_tensorcnt(0);
  }
  __syncthreads();
  for (int i = tid; i < 128*HID; i += 256) {          /* h half: broadcast hidden */
    int t = i >> 6, j = i & 63;
    s_c[t][HID + j] = hid16[b*HID + j];
  }
  __syncthreads();
  const int wave = tid >> 5, lane = tid & 31;
  const int tt = wave*16;                             /* Linear 128->64, WMMA K=128 */
#pragma unroll
  for (int nt = 0; nt < 4; ++nt) {
    v8f acc = acc_bias(b1, nt*16);
    acc = gemm16(&s_c[0][0], 2*HID, tt, w1p, nt, 4, acc);
    int n  = nt*16 + (lane & 15);
    int tb = tt + ((lane >> 4) << 3);
#pragma unroll
    for (int v = 0; v < 8; ++v) s_h[tb + v][n] = fmaxf(acc[v], 0.0f);
  }
  __syncthreads();
  if (tid < 128) {                                    /* Linear 64->1 */
    float sum = b2[0];
    for (int j = 0; j < HID; ++j) sum += s_h[tid][j] * w2[j];
    out[t0 + tid] = sum;
  }
}

/* ============================ launcher ============================ */
extern "C" void kernel_launch(void* const* d_in, const int* in_sizes, int n_in,
                              void* d_out, int out_size, void* d_ws, size_t ws_size,
                              hipStream_t stream) {
  (void)in_sizes; (void)n_in; (void)out_size; (void)ws_size;
  const float* inp      = (const float*)d_in[0];
  const float* h0       = (const float*)d_in[1];
  const float* enc_w1   = (const float*)d_in[2];
  const float* enc_b1   = (const float*)d_in[3];
  const float* enc_g    = (const float*)d_in[4];
  const float* enc_beta = (const float*)d_in[5];
  const float* enc_w2   = (const float*)d_in[6];
  const float* enc_b2   = (const float*)d_in[7];
  const float* l_w1     = (const float*)d_in[8];
  const float* l_b1     = (const float*)d_in[9];
  const float* l_g      = (const float*)d_in[10];
  const float* l_beta   = (const float*)d_in[11];
  const float* l_w2     = (const float*)d_in[12];
  const float* l_b2     = (const float*)d_in[13];
  const float* l_gwih   = (const float*)d_in[14];
  const float* l_gwhh   = (const float*)d_in[15];
  const float* l_gbih   = (const float*)d_in[16];
  const float* l_gbhh   = (const float*)d_in[17];
  const float* g_wih    = (const float*)d_in[18];
  const float* g_whh    = (const float*)d_in[19];
  const float* g_bih    = (const float*)d_in[20];
  const float* g_bhh    = (const float*)d_in[21];
  const float* head_w1  = (const float*)d_in[22];
  const float* head_b1  = (const float*)d_in[23];
  const float* head_w2  = (const float*)d_in[24];
  const float* head_b2  = (const float*)d_in[25];

  _Float16* wsh = (_Float16*)d_ws;
  _Float16* xg  = wsh + O_X;
  float* wsf  = (float*)(wsh + H_TOTAL);
  float* part = wsf + F_PART;
  float* gmax = wsf + F_GMAX;
  float* newh = wsf + F_NEWH;
  float* out  = (float*)d_out;

  k_prep<<<dim3((143360 + 255)/256), dim3(256), 0, stream>>>(
      enc_w2, l_w1, l_w2, l_gwih, l_gwhh, head_w1, wsh);

  k_encoder<<<dim3(NTOK/128), dim3(256), 0, stream>>>(
      inp, enc_w1, enc_b1, enc_g, enc_beta, wsh + O_ENCW2, enc_b2, xg);

  for (int l = 0; l < 2; ++l) {
    {
      const int ps = l*2 + 0;     /* row pass */
      k_pass<128, 1><<<dim3(NB*NM), dim3(256), 0, stream>>>(
          xg, wsh + O_LW1 + ps*4096, wsh + O_LW2 + ps*4096,
          wsh + O_GWIH + ps*12288, wsh + O_GWHH + ps*12288,
          l_b1 + ps*64, l_g + ps*64, l_beta + ps*64, l_b2 + ps*64,
          l_gbih + ps*192, l_gbhh + ps*192);
    }
    {
      const int ps = l*2 + 1;     /* col pass */
      k_pass<256, NK><<<dim3(NB*NK), dim3(256), 0, stream>>>(
          xg, wsh + O_LW1 + ps*4096, wsh + O_LW2 + ps*4096,
          wsh + O_GWIH + ps*12288, wsh + O_GWHH + ps*12288,
          l_b1 + ps*64, l_g + ps*64, l_beta + ps*64, l_b2 + ps*64,
          l_gbih + ps*192, l_gbhh + ps*192);
    }
  }

  k_gmax1<<<dim3(NB*64), dim3(256), 0, stream>>>(xg, part);
  k_gmax2<<<dim3(NB), dim3(64), 0, stream>>>(part, gmax);

  k_topgru<<<dim3((NB*HID + 255)/256), dim3(256), 0, stream>>>(
      gmax, h0, g_wih, g_whh, g_bih, g_bhh, newh, wsh + O_HID16, out + NTOK);

  k_head<<<dim3(NTOK/128), dim3(256), 0, stream>>>(
      xg, wsh + O_HID16, wsh + O_HEADW1, head_b1, head_w2, head_b2, out);
}
